// MultiHeadAttention_87136296501294
// MI455X (gfx1250) — matrix operand order repaired and checked by reading, not compiled, not run on hardware
//
#include <hip/hip_runtime.h>
#include <hip/hip_bf16.h>

typedef __attribute__((ext_vector_type(16))) _Float16 v16h;
typedef __attribute__((ext_vector_type(8)))  _Float16 v8h;
typedef __attribute__((ext_vector_type(8)))  float    v8f;

#define D_MODEL 256
#define SEQ     4096
#define BATCH   2
#define NHEAD   8
#define DKH     32
#define LN_EPS  1e-5f

// ---------------------------------------------------------------------------
// WMMA helpers (CDNA5 wave32, V_WMMA_F32_16X16X32_F16)
// ---------------------------------------------------------------------------
__device__ __forceinline__ v8f wmma_f16(v16h a, v16h b, v8f c) {
  return __builtin_amdgcn_wmma_f32_16x16x32_f16(
      /*neg_a=*/false, a, /*neg_b=*/false, b,
      /*c_mod=*/(short)0, c, /*reuse_a=*/false, /*reuse_b=*/false);
}

// A operand: 16(M) x 32(K) f16 tile, row-major, row stride ld (in halves).
// Lanes 0-15: row M=lane, K = 0..7 and 16..23. Lanes 16-31: K = 8..15, 24..31.
__device__ __forceinline__ v16h load_a_rm(const _Float16* tile, int ld) {
  const unsigned lane = threadIdx.x & 31u;
  const unsigned m    = lane & 15u;
  const unsigned kb   = (lane >> 4) << 3;           // 0 or 8
  const _Float16* r   = tile + (size_t)m * ld + kb;
  v8h lo = *(const v8h*)(r);                        // K = kb .. kb+7
  v8h hi = *(const v8h*)(r + 16);                   // K = kb+16 .. kb+23
  v16h a;
#pragma unroll
  for (int i = 0; i < 8; ++i) { a[i] = lo[i]; a[i + 8] = hi[i]; }
  return a;
}

// B operand: 32(K) x 16(N), where B[k][n] = tile[n*ld + k]  (transposed read,
// i.e. tile is row-major [N][K] like a weight matrix W[n][k]).
__device__ __forceinline__ v16h load_b_tr(const _Float16* tile, int ld) {
  const unsigned lane = threadIdx.x & 31u;
  const unsigned n    = lane & 15u;
  const unsigned kb   = (lane >> 4) << 3;           // 0 or 8
  const _Float16* r   = tile + (size_t)n * ld + kb;
  v8h lo = *(const v8h*)(r);
  v8h hi = *(const v8h*)(r + 16);
  v16h b;
#pragma unroll
  for (int i = 0; i < 8; ++i) { b[i] = lo[i]; b[i + 8] = hi[i]; }
  return b;
}

// ---------------------------------------------------------------------------
// Kernel 1: transpose x [B][D][S] -> xt f16 [B][S][D]; weights f32 -> f16
// ---------------------------------------------------------------------------
__global__ void prep_kernel(const float* __restrict__ x,
                            const float* __restrict__ wq,
                            const float* __restrict__ wk,
                            const float* __restrict__ wv,
                            const float* __restrict__ wo,
                            _Float16* __restrict__ xtf,
                            _Float16* __restrict__ wqf,
                            _Float16* __restrict__ wkf,
                            _Float16* __restrict__ wvf,
                            _Float16* __restrict__ wof) {
  const size_t NXT = (size_t)BATCH * SEQ * D_MODEL;   // 2,097,152
  size_t i = (size_t)blockIdx.x * blockDim.x + threadIdx.x;
  if (i < NXT) {
    unsigned d = (unsigned)(i & 255u);
    unsigned s = (unsigned)((i >> 8) & 4095u);
    unsigned b = (unsigned)(i >> 20);
    xtf[i] = (_Float16)x[((size_t)b * D_MODEL + d) * SEQ + s];
  } else {
    size_t j = i - NXT;
    if      (j <  65536) wqf[j]          = (_Float16)wq[j];
    else if (j < 131072) wkf[j - 65536]  = (_Float16)wk[j - 65536];
    else if (j < 196608) wvf[j - 131072] = (_Float16)wv[j - 131072];
    else if (j < 262144) wof[j - 196608] = (_Float16)wo[j - 196608];
  }
}

// ---------------------------------------------------------------------------
// Kernel 2: Q/K/V = xt @ W^T + b.
//   Q, K stored per-head row-major  [B][H][S][DK]   (A / B^T operands later)
//   V stored per-head TRANSPOSED    [B][H][DK][S]   (A operand of V^T · P^T)
// One wave per 16x16 output tile, K-loop of 8 WMMAs.
// ---------------------------------------------------------------------------
__global__ void qkv_kernel(const _Float16* __restrict__ xtf,
                           const _Float16* __restrict__ wqf,
                           const _Float16* __restrict__ wkf,
                           const _Float16* __restrict__ wvf,
                           const float* __restrict__ bq,
                           const float* __restrict__ bk,
                           const float* __restrict__ bv,
                           _Float16* __restrict__ qbuf,
                           _Float16* __restrict__ kbuf,
                           _Float16* __restrict__ vtbuf) {
  const unsigned wid  = blockIdx.x * 8u + (threadIdx.x >> 5);
  const unsigned lane = threadIdx.x & 31u;
  const unsigned mat  = wid >> 13;                  // 0,1,2 -> q,k,v
  const unsigned t    = wid & 8191u;
  const unsigned m0   = (t >> 4) << 4;              // 0..8176
  const unsigned n0   = (t & 15u) << 4;             // 0..240

  const _Float16* w   = (mat == 0) ? wqf : (mat == 1) ? wkf : wvf;
  const float*    bia = (mat == 0) ? bq  : (mat == 1) ? bk  : bv;

  v8f c = {};
#pragma unroll
  for (int k0 = 0; k0 < D_MODEL; k0 += 32) {
    v16h a = load_a_rm(xtf + (size_t)m0 * D_MODEL + k0, D_MODEL);
    v16h b = load_b_tr(w   + (size_t)n0 * D_MODEL + k0, D_MODEL);
    c = wmma_f16(a, b, c);
  }

  const unsigned n  = n0 + (lane & 15u);
  const unsigned h  = n >> 5;
  const unsigned dk = n & 31u;
  const float    bn = bia[n];
  const unsigned mh = (lane >> 4) << 3;             // 0 or 8
  if (mat < 2) {
    _Float16* dst = (mat == 0) ? qbuf : kbuf;
#pragma unroll
    for (int r = 0; r < 8; ++r) {
      unsigned m = m0 + r + mh;
      unsigned b = m >> 12;                         // / SEQ
      unsigned s = m & 4095u;
      dst[(((size_t)b * NHEAD + h) * SEQ + s) * DKH + dk] = (_Float16)(c[r] + bn);
    }
  } else {
#pragma unroll
    for (int r = 0; r < 8; ++r) {
      unsigned m = m0 + r + mh;
      unsigned b = m >> 12;
      unsigned s = m & 4095u;
      vtbuf[(((size_t)b * NHEAD + h) * DKH + dk) * SEQ + s] = (_Float16)(c[r] + bn);
    }
  }
}

// ---------------------------------------------------------------------------
// Kernel 3: flash attention, fully transposed dataflow.
//   S^T   = K · Q^T      (A = 16-key block of K, B = Q^T; D: M=key, N=query)
//   ctx^T = V^T · P^T    (A = V^T dk-tile,       B = P^T from LDS)
// Queries live along lanes -> softmax stats are scalars per lane; the row
// reduction is a local 8-reg reduce + ONE shfl_xor(16) with the partner lane,
// and the alpha rescale applies per-lane uniformly to the accumulators.
// P^T staging: each lane's keys are contiguous -> 2x ds_store_b128 +
// 2x ds_load_b128 per step (DS ops of one wave complete in order).
// ---------------------------------------------------------------------------
__global__ void attn_kernel(const _Float16* __restrict__ qbuf,
                            const _Float16* __restrict__ kbuf,
                            const _Float16* __restrict__ vtbuf,
                            _Float16* __restrict__ ctxbuf) {
  __shared__ _Float16 pstage[8][16 * 32];           // 8 waves * 1KB

  const unsigned wave = threadIdx.x >> 5;
  const unsigned lane = threadIdx.x & 31u;
  const unsigned tile = blockIdx.x * 8u + wave;     // 0..4095
  const unsigned qt   = tile & 255u;                // query tile in sequence
  const unsigned bh   = tile >> 8;                  // b*NHEAD + h
  const unsigned q0   = qt << 4;

  const _Float16* qh = qbuf  + (size_t)bh * SEQ * DKH;
  const _Float16* kh = kbuf  + (size_t)bh * SEQ * DKH;
  const _Float16* vt = vtbuf + (size_t)bh * DKH * SEQ;
  _Float16* myp = pstage[wave];

  // B operand: Q^T (32 dk x 16 queries), pre-scaled by 1/sqrt(dk)
  v16h qb = load_b_tr(qh + (size_t)q0 * DKH, DKH);
#pragma unroll
  for (int i = 0; i < 16; ++i)
    qb[i] = (_Float16)((float)qb[i] * 0.17677669529663687f);

  v8f acc0 = {}, acc1 = {};                         // ctx^T: dk 0-15 / 16-31
  float mrun = -1e30f, lrun = 0.0f;
  const unsigned nl = lane & 15u;                   // this lane's query
  const unsigned g8 = (lane >> 4) << 3;             // key/dk sub-offset: 0 or 8

  for (int j0 = 0; j0 < SEQ; j0 += 32) {
    // prefetch two blocks ahead (stays inside the workspace)
    __builtin_prefetch(kh + (size_t)(j0 + 64) * DKH, 0, 1);
    __builtin_prefetch(vt + (size_t)(j0 + 64), 0, 1);

    v16h ka0 = load_a_rm(kh + (size_t)j0 * DKH, DKH);          // keys j0..+15
    v16h ka1 = load_a_rm(kh + (size_t)(j0 + 16) * DKH, DKH);   // keys +16..+31
    v8f z = {};
    v8f s0 = wmma_f16(ka0, qb, z);   // reg r: key j0 + g8 + r      , query nl
    v8f s1 = wmma_f16(ka1, qb, z);   // reg r: key j0 + 16 + g8 + r , query nl

    // online softmax: 16 local values + one partner-lane combine
    float mx = fmaxf(s0[0], s1[0]);
#pragma unroll
    for (int r = 1; r < 8; ++r) mx = fmaxf(mx, fmaxf(s0[r], s1[r]));
    mx = fmaxf(mx, __shfl_xor(mx, 16, 32));
    const float mnew  = fmaxf(mrun, mx);
    const float alpha = __expf(mrun - mnew);
    float rs = 0.0f;
    v8h p0, p1;
#pragma unroll
    for (int r = 0; r < 8; ++r) {
      float e0 = __expf(s0[r] - mnew);
      float e1 = __expf(s1[r] - mnew);
      rs += e0 + e1;
      p0[r] = (_Float16)e0;
      p1[r] = (_Float16)e1;
    }
    rs += __shfl_xor(rs, 16, 32);
    lrun = lrun * alpha + rs;
    mrun = mnew;
#pragma unroll
    for (int r = 0; r < 8; ++r) { acc0[r] *= alpha; acc1[r] *= alpha; }

    // P^T -> LDS [query][key]: this lane's keys are contiguous
    *(v8h*)(myp + nl * 32 + g8)      = p0;          // keys g8..g8+7
    *(v8h*)(myp + nl * 32 + 16 + g8) = p1;          // keys 16+g8..16+g8+7

    // ctx^T += V^T · P^T
    v16h pb  = load_b_tr(myp, 32);                  // 32 keys x 16 queries
    v16h va0 = load_a_rm(vt + j0, SEQ);             // dk 0..15  x 32 keys
    v16h va1 = load_a_rm(vt + (size_t)16 * SEQ + j0, SEQ);  // dk 16..31
    acc0 = wmma_f16(va0, pb, acc0);
    acc1 = wmma_f16(va1, pb, acc1);
  }

  // normalize; store ctx as f16 [B][S][256] (row = query, packed b128 stores)
  const unsigned b = bh >> 3;
  const unsigned h = bh & 7u;
  const float  inv = 1.0f / lrun;
  v8h o0, o1;
#pragma unroll
  for (int r = 0; r < 8; ++r) {
    o0[r] = (_Float16)(acc0[r] * inv);              // dk = g8 + r
    o1[r] = (_Float16)(acc1[r] * inv);              // dk = 16 + g8 + r
  }
  _Float16* dst = ctxbuf + ((size_t)b * SEQ + q0 + nl) * D_MODEL + h * DKH + g8;
  *(v8h*)(dst)      = o0;
  *(v8h*)(dst + 16) = o1;
}

// ---------------------------------------------------------------------------
// Kernel 4: out = ctx @ Wo^T + b_o + residual; LayerNorm over D; store [B][C][S]
// One block per 16 rows; 8 waves x 2 N-tiles cover all 256 columns.
// ---------------------------------------------------------------------------
__global__ void oproj_ln_kernel(const _Float16* __restrict__ ctxbuf,
                                const _Float16* __restrict__ wof,
                                const float* __restrict__ bo,
                                const float* __restrict__ x,
                                const float* __restrict__ gamma,
                                const float* __restrict__ beta,
                                float* __restrict__ out) {
  __shared__ float ybuf[16][260];                   // pad to dodge bank conflicts

  const unsigned wave = threadIdx.x >> 5;
  const unsigned lane = threadIdx.x & 31u;
  const unsigned m0   = blockIdx.x << 4;
  const unsigned n0   = wave * 32u;

  v8f c0 = {}, c1 = {};
#pragma unroll
  for (int k0 = 0; k0 < D_MODEL; k0 += 32) {
    v16h a  = load_a_rm(ctxbuf + (size_t)m0 * D_MODEL + k0, D_MODEL);
    v16h b0 = load_b_tr(wof + (size_t)n0 * D_MODEL + k0, D_MODEL);
    v16h b1 = load_b_tr(wof + (size_t)(n0 + 16) * D_MODEL + k0, D_MODEL);
    c0 = wmma_f16(a, b0, c0);
    c1 = wmma_f16(a, b1, c1);
  }

  const unsigned mh = (lane >> 4) << 3;
  const unsigned nl = lane & 15u;
  const float bn0 = bo[n0 + nl];
  const float bn1 = bo[n0 + 16 + nl];
#pragma unroll
  for (int r = 0; r < 8; ++r) {
    unsigned m = r + mh;
    ybuf[m][n0 + nl]      = c0[r] + bn0;
    ybuf[m][n0 + 16 + nl] = c1[r] + bn1;
  }
  __syncthreads();

  // LayerNorm: 16 threads per row, 16 channels per thread
  const unsigned row = threadIdx.x >> 4;
  const unsigned cb  = threadIdx.x & 15u;
  const unsigned m   = m0 + row;
  const unsigned b   = m >> 12;
  const unsigned s   = m & 4095u;
  const float* xcol  = x + (size_t)b * D_MODEL * SEQ + s;   // stride SEQ per channel

  float vals[16];
  float s1 = 0.0f, s2 = 0.0f;
#pragma unroll
  for (int i = 0; i < 16; ++i) {
    unsigned d = cb + i * 16u;
    float y = ybuf[row][d] + xcol[(size_t)d * SEQ];         // + residual
    vals[i] = y;
    s1 += y;
    s2 += y * y;
  }
#pragma unroll
  for (int msk = 1; msk < 16; msk <<= 1) {
    s1 += __shfl_xor(s1, msk, 32);
    s2 += __shfl_xor(s2, msk, 32);
  }
  const float mean = s1 * (1.0f / D_MODEL);
  const float var  = s2 * (1.0f / D_MODEL) - mean * mean;
  const float rstd = rsqrtf(var + LN_EPS);
#pragma unroll
  for (int i = 0; i < 16; ++i) {
    unsigned d = cb + i * 16u;
    out[((size_t)b * D_MODEL + d) * SEQ + s] =
        (vals[i] - mean) * rstd * gamma[d] + beta[d];
  }
}

// ---------------------------------------------------------------------------
// Launch
// ---------------------------------------------------------------------------
extern "C" void kernel_launch(void* const* d_in, const int* in_sizes, int n_in,
                              void* d_out, int out_size, void* d_ws, size_t ws_size,
                              hipStream_t stream) {
  const float* x     = (const float*)d_in[0];
  const float* w_q   = (const float*)d_in[1];
  const float* b_q   = (const float*)d_in[2];
  const float* w_k   = (const float*)d_in[3];
  const float* b_k   = (const float*)d_in[4];
  const float* w_v   = (const float*)d_in[5];
  const float* b_v   = (const float*)d_in[6];
  const float* w_o   = (const float*)d_in[7];
  const float* b_o   = (const float*)d_in[8];
  const float* gamma = (const float*)d_in[9];
  const float* beta  = (const float*)d_in[10];
  float* out = (float*)d_out;

  // workspace layout (bytes)
  char* ws = (char*)d_ws;
  const size_t XT_SZ = (size_t)BATCH * SEQ * D_MODEL * sizeof(_Float16);  // 4 MB
  const size_t W_SZ  = (size_t)D_MODEL * D_MODEL * sizeof(_Float16);      // 128 KB
  _Float16* xtf  = (_Float16*)(ws);
  _Float16* wqf  = (_Float16*)(ws + XT_SZ);
  _Float16* wkf  = (_Float16*)(ws + XT_SZ + W_SZ);
  _Float16* wvf  = (_Float16*)(ws + XT_SZ + 2 * W_SZ);
  _Float16* wof  = (_Float16*)(ws + XT_SZ + 3 * W_SZ);
  _Float16* qbuf = (_Float16*)(ws + XT_SZ + 4 * W_SZ);
  _Float16* kbuf = (_Float16*)(ws + 2 * XT_SZ + 4 * W_SZ);
  _Float16* vtb  = (_Float16*)(ws + 3 * XT_SZ + 4 * W_SZ);
  _Float16* ctxb = (_Float16*)(ws + 4 * XT_SZ + 4 * W_SZ);

  // 1) transpose + f16 conversion: 2,097,152 + 262,144 elements
  {
    const size_t total = (size_t)BATCH * SEQ * D_MODEL + 4 * (size_t)D_MODEL * D_MODEL;
    dim3 grid((unsigned)((total + 255) / 256));
    prep_kernel<<<grid, 256, 0, stream>>>(x, w_q, w_k, w_v, w_o,
                                          xtf, wqf, wkf, wvf, wof);
  }
  // 2) QKV projections: 3 * 512 * 16 tiles, 8 waves/block
  {
    dim3 grid(3 * 512 * 16 / 8);
    qkv_kernel<<<grid, 256, 0, stream>>>(xtf, wqf, wkf, wvf, b_q, b_k, b_v,
                                         qbuf, kbuf, vtb);
  }
  // 3) flash attention: B*H*(S/16) = 4096 waves, 8 waves/block
  {
    dim3 grid(4096 / 8);
    attn_kernel<<<grid, 256, 0, stream>>>(qbuf, kbuf, vtb, ctxb);
  }
  // 4) output projection + residual + LayerNorm: 8192/16 row-blocks
  {
    dim3 grid((BATCH * SEQ) / 16);
    oproj_ln_kernel<<<grid, 256, 0, stream>>>(ctxb, wof, b_o, x, gamma, beta, out);
  }
}
